// FastRx_wo_Diag_19473381720179
// MI455X (gfx1250) — matrix operand order repaired and checked by reading, not compiled, not run on hardware
//
#include <hip/hip_runtime.h>
#include <math.h>

// ---------------------------------------------------------------------------
// FastRx (wo diag) pipeline for MI455X / gfx1250.
// Heavy work = GCN adjacency GEMMs (M=4000,K=4000,N=256) -> bf16 WMMA path.
// A is pre-converted to bf16 row-major (one cvt per element, not per use);
// B is pre-converted + pre-swizzled to WMMA fragment order. The GEMM inner
// loop is 2x b128 (A) + 8x b128 (B) + 4x v_wmma per K-step, ping-pong
// double-buffered (unroll-2, no register rotation) so each compute stage
// waits only on loads issued one full stage earlier.
// ---------------------------------------------------------------------------

typedef __attribute__((ext_vector_type(16))) __bf16 bf16x16;
typedef __attribute__((ext_vector_type(8)))  __bf16 bf16x8;
typedef __attribute__((ext_vector_type(4)))  __bf16 bf16x4;
typedef __attribute__((ext_vector_type(8)))  float  f32x8;

__device__ __forceinline__ float wred_sum(float v) {
#pragma unroll
  for (int o = 16; o > 0; o >>= 1) v += __shfl_xor(v, o, 32);
  return v;
}
__device__ __forceinline__ float wred_max(float v) {
#pragma unroll
  for (int o = 16; o > 0; o >>= 1) v = fmaxf(v, __shfl_xor(v, o, 32));
  return v;
}

// ---------------------------------------------------------------------------
// K1: embedding gather + mean over 32 codes + width-3 conv over features + ReLU
// grid(64 visits), block(128 features)
// ---------------------------------------------------------------------------
__global__ __launch_bounds__(128)
void embed_conv_kernel(const int* __restrict__ codes,
                       const float* __restrict__ emb,
                       const float* __restrict__ cw,
                       const float* __restrict__ cb,
                       float* __restrict__ i2) {
  const int vi = blockIdx.x;
  const int f  = threadIdx.x;
  float m = 0.f;
#pragma unroll 4
  for (int p = 0; p < 32; ++p)
    m += emb[(size_t)codes[vi * 32 + p] * 128 + f];
  m *= (1.0f / 32.0f);
  __shared__ float ms[128];
  ms[f] = m;
  __syncthreads();
  const float left  = (f > 0)   ? ms[f - 1] : 0.f;
  const float right = (f < 127) ? ms[f + 1] : 0.f;
  i2[vi * 128 + f] = fmaxf(cw[0] * left + cw[1] * m + cw[2] * right + cb[0], 0.f);
}

// ---------------------------------------------------------------------------
// K2a: q/k/v = x @ {wq,wk,wv}   (64x128 @ 128x256)
// grid(3*64), block(256)
// ---------------------------------------------------------------------------
__global__ __launch_bounds__(256)
void qkv_kernel(const float* __restrict__ x,
                const float* __restrict__ wq,
                const float* __restrict__ wk,
                const float* __restrict__ wv,
                float* __restrict__ q, float* __restrict__ k, float* __restrict__ v) {
  const int mat = blockIdx.x >> 6;
  const int row = blockIdx.x & 63;
  const int c   = threadIdx.x;
  __shared__ float xs[128];
  if (c < 128) xs[c] = x[row * 128 + c];
  __syncthreads();
  const float* W = (mat == 0) ? wq : ((mat == 1) ? wk : wv);
  float*       O = (mat == 0) ? q  : ((mat == 1) ? k  : v);
  float s = 0.f;
#pragma unroll 4
  for (int d = 0; d < 128; ++d) s += xs[d] * W[d * 256 + c];
  O[row * 256 + c] = s;
}

// ---------------------------------------------------------------------------
// K2b: fastformer gates. gq[c] = sum_i q[i,c]*softmax_c(q*alpha*s)[i,c]
//      p = gq*k ; gk[c] = sum_i p[i,c]*softmax_c(p*beta*s)[i,c]
// single block, 8 waves, fixed-order reductions (deterministic).
// ---------------------------------------------------------------------------
__global__ __launch_bounds__(256)
void fastformer_gates(const float* __restrict__ qg,
                      const float* __restrict__ kg,
                      const float* __restrict__ alpha,
                      const float* __restrict__ beta,
                      float* __restrict__ gk_out) {
  const int tid  = threadIdx.x;
  const int w    = tid >> 5;
  const int lane = tid & 31;
  const float scale = 0.0625f; // 256^-0.5
  __shared__ float part[8][256];
  __shared__ float gq_s[256];

  // ---- stage 1: gq ----
#pragma unroll
  for (int u = 0; u < 8; ++u) part[w][u * 32 + lane] = 0.f;
  __syncthreads();
  for (int i = w; i < 64; i += 8) {
    float qv[8], tl[8];
#pragma unroll
    for (int u = 0; u < 8; ++u) {
      const int c = u * 32 + lane;
      qv[u] = qg[i * 256 + c];
      tl[u] = qv[u] * alpha[c] * scale;
    }
    float mx = -3.4e38f;
#pragma unroll
    for (int u = 0; u < 8; ++u) mx = fmaxf(mx, tl[u]);
    mx = wred_max(mx);
    float se = 0.f;
#pragma unroll
    for (int u = 0; u < 8; ++u) se += __expf(tl[u] - mx);
    se = wred_sum(se);
#pragma unroll
    for (int u = 0; u < 8; ++u) {
      const int c = u * 32 + lane;
      part[w][c] += qv[u] * (__expf(tl[u] - mx) / se);
    }
  }
  __syncthreads();
  if (tid < 256) {
    float g = 0.f;
#pragma unroll
    for (int ww = 0; ww < 8; ++ww) g += part[ww][tid];
    gq_s[tid] = g;
  }
  __syncthreads();

  // ---- stage 2: gk ----
#pragma unroll
  for (int u = 0; u < 8; ++u) part[w][u * 32 + lane] = 0.f;
  __syncthreads();
  for (int i = w; i < 64; i += 8) {
    float pv[8], tl[8];
#pragma unroll
    for (int u = 0; u < 8; ++u) {
      const int c = u * 32 + lane;
      pv[u] = gq_s[c] * kg[i * 256 + c];
      tl[u] = pv[u] * beta[c] * scale;
    }
    float mx = -3.4e38f;
#pragma unroll
    for (int u = 0; u < 8; ++u) mx = fmaxf(mx, tl[u]);
    mx = wred_max(mx);
    float se = 0.f;
#pragma unroll
    for (int u = 0; u < 8; ++u) se += __expf(tl[u] - mx);
    se = wred_sum(se);
#pragma unroll
    for (int u = 0; u < 8; ++u) {
      const int c = u * 32 + lane;
      part[w][c] += pv[u] * (__expf(tl[u] - mx) / se);
    }
  }
  __syncthreads();
  if (tid < 256) {
    float g = 0.f;
#pragma unroll
    for (int ww = 0; ww < 8; ++ww) g += part[ww][tid];
    gk_out[tid] = g;
  }
}

// ---------------------------------------------------------------------------
// K2c: feat[i,c] = (gk .* v[i]) @ wr[:,c] + q[i,c]
// grid(64), block(256)
// ---------------------------------------------------------------------------
__global__ __launch_bounds__(256)
void feat_combine(const float* __restrict__ q,
                  const float* __restrict__ v,
                  const float* __restrict__ gk,
                  const float* __restrict__ wr,
                  float* __restrict__ feat) {
  const int i = blockIdx.x;
  const int c = threadIdx.x;
  __shared__ float gv[256];
  gv[c] = gk[c] * v[i * 256 + c];
  __syncthreads();
  float s = 0.f;
#pragma unroll 4
  for (int d = 0; d < 256; ++d) s += gv[d] * wr[d * 256 + c];
  feat[i * 256 + c] = s + q[i * 256 + c];
}

// ---------------------------------------------------------------------------
// K3a: elementwise f32 -> bf16 (row-major copy). n % 4 == 0.
// ---------------------------------------------------------------------------
__global__ __launch_bounds__(256)
void convert_bf16(const float* __restrict__ in, int n, __bf16* __restrict__ out) {
  const int t = (blockIdx.x * 256 + threadIdx.x) * 4;
  if (t < n) {
    const float4 f = *(const float4*)(in + t);
    bf16x4 o;
    o[0] = (__bf16)f.x; o[1] = (__bf16)f.y; o[2] = (__bf16)f.z; o[3] = (__bf16)f.w;
    *(bf16x4*)(out + t) = o;
  }
}

// ---------------------------------------------------------------------------
// K3b: pack B (K x 256, f32 row-major) into bf16 WMMA-fragment order:
//      Bpack[((k/32)*256 + c)*32 + (k%32)] = bf16(B[k*256+c])
// grid(K), block(256) -> one k-row per block
// ---------------------------------------------------------------------------
__global__ __launch_bounds__(256)
void pack_b_bf16(const float* __restrict__ B, int K, __bf16* __restrict__ out) {
  const int k = blockIdx.x;
  const int c = threadIdx.x;
  out[((size_t)(k >> 5) * 256 + c) * 32 + (k & 31)] = (__bf16)B[(size_t)k * 256 + c];
}

// ---------------------------------------------------------------------------
// K3c: GEMM fragment helpers (ping-pong stages)
// ---------------------------------------------------------------------------
struct Frag {
  bf16x8 a0, a1;
  bf16x8 b0[4], b1[4];
};

__device__ __forceinline__ void load_frag(Frag& f,
                                          const __bf16* __restrict__ arow,
                                          const __bf16* __restrict__ bbase,
                                          int panel) {
  const __bf16* ap = arow + (panel << 5);
  f.a0 = *(const bf16x8*)(ap);
  f.a1 = *(const bf16x8*)(ap + 8);
  const __bf16* bp = bbase + (size_t)panel * 8192;
#pragma unroll
  for (int sub = 0; sub < 4; ++sub) {
    f.b0[sub] = *(const bf16x8*)(bp + sub * 512);
    f.b1[sub] = *(const bf16x8*)(bp + sub * 512 + 8);
  }
}

__device__ __forceinline__ void compute_frag(const Frag& f, f32x8 acc[4]) {
  const bf16x16 a = __builtin_shufflevector(
      f.a0, f.a1, 0, 1, 2, 3, 4, 5, 6, 7, 8, 9, 10, 11, 12, 13, 14, 15);
#pragma unroll
  for (int sub = 0; sub < 4; ++sub) {
    const bf16x16 b = __builtin_shufflevector(
        f.b0[sub], f.b1[sub], 0, 1, 2, 3, 4, 5, 6, 7, 8, 9, 10, 11, 12, 13, 14, 15);
    acc[sub] = __builtin_amdgcn_wmma_f32_16x16x32_bf16(
        false, a, false, b, (short)0, acc[sub], false, false);
  }
}

// ---------------------------------------------------------------------------
// K3d: C[M,256] = act(A[M,K] @ B[K,256] + bias) * scale (+ C_prev)
// bf16 WMMA 16x16x32, f32 accumulate. Wave tile = 16(M) x 64(N).
// A: bf16 row-major (pre-converted) -> 2x global_load_b128 per K-step.
// B: pre-packed bf16 fragments     -> 2x global_load_b128 per sub-tile.
// Unroll-2 ping-pong double buffering: compute(S) waits only on loads
// issued a full stage earlier; no register rotation movs.
// M%16==0, K%32==0, N==256. Wave-uniform control flow (EXEC all-1s, ISA 7.12).
// ---------------------------------------------------------------------------
__global__ __launch_bounds__(256)
void gemm_wmma_n256(const __bf16* __restrict__ A, int lda,
                    const __bf16* __restrict__ Bpack,
                    float* __restrict__ C,
                    int M, int K,
                    const float* __restrict__ bias,
                    int do_relu,
                    const float* __restrict__ scale_ptr, float scale_mul,
                    int accumulate) {
  const int gw     = (blockIdx.x * blockDim.x + threadIdx.x) >> 5;
  const int lane   = threadIdx.x & 31;
  const int tile_m = gw >> 2;      // M/16 tiles
  const int tile_n = gw & 3;       // 4 tiles of 64 cols
  if (tile_m * 16 >= M) return;    // wave-uniform

  const int row0  = tile_m * 16;
  const int col0  = tile_n * 64;
  const int mrow  = lane & 15;     // A row / B col / C col within tile
  const int khalf = lane >> 4;     // K-half select (this code's contiguous k = 16*khalf + s)

  f32x8 acc[4] = {};
  // A fragment 16x32: lanes 0-15 K {kk..kk+15},
  //                   lanes 16-31 K {kk+16..kk+31}  (K-contiguous x16)
  const __bf16* arow  = A + (size_t)(row0 + mrow) * lda + khalf * 16;
  // B fragment: lane reads 16 consecutive bf16 at its (col, khalf) slot
  const __bf16* bbase = Bpack + (size_t)(col0 + mrow) * 32 + khalf * 16;

  const int npanels = K >> 5;
  Frag s0, s1;
  load_frag(s0, arow, bbase, 0);

  int p = 0;
  for (; p + 1 < npanels; p += 2) {
    load_frag(s1, arow, bbase, p + 1);
    compute_frag(s0, acc);
    const int pn = (p + 2 < npanels) ? p + 2 : p;  // clamp: redundant reload ok
    load_frag(s0, arow, bbase, pn);
    compute_frag(s1, acc);
  }
  if (p < npanels) compute_frag(s0, acc);  // odd panel count tail (125 panels)

  float s = scale_mul;
  if (scale_ptr) s *= scale_ptr[0];
#pragma unroll
  for (int sub = 0; sub < 4; ++sub) {
    const int c = col0 + sub * 16 + mrow;
#pragma unroll
    for (int vv = 0; vv < 8; ++vv) {
      const int r = row0 + khalf * 8 + vv;
      float t = acc[sub][vv];
      if (bias) t += bias[c];
      if (do_relu) t = fmaxf(t, 0.f);
      t *= s;
      const size_t idx = (size_t)r * 256 + c;
      if (accumulate) t += C[idx];
      C[idx] = t;
    }
  }
}

// ---------------------------------------------------------------------------
// K5a: decoder head -> hidden (512). Single block, deterministic reductions.
// ---------------------------------------------------------------------------
__global__ __launch_bounds__(256)
void decode_head(const float* __restrict__ feat,
                 const float* __restrict__ dm,
                 const int* __restrict__ med_codes,
                 const float* __restrict__ w1,
                 const float* __restrict__ b1,
                 float* __restrict__ hidden) {
  const int tid  = threadIdx.x;
  const int w    = tid >> 5;
  const int lane = tid & 31;
  __shared__ float qs[256], f1[256], f2[256];
  __shared__ float l1[4000];
  __shared__ float l2[64];
  __shared__ float red[256];
  __shared__ float agg[4000];
  __shared__ int   cds[1008]; // 63 visits x 16 codes

  qs[tid] = feat[63 * 256 + tid];
  for (int t = tid; t < 1008; t += 256) cds[t] = med_codes[t];
  __syncthreads();

  // logits1[j] = query . dm[j]
  for (int j = w; j < 4000; j += 8) {
    float s = 0.f;
    for (int d = lane; d < 256; d += 32) s += qs[d] * dm[(size_t)j * 256 + d];
    s = wred_sum(s);
    if (lane == 0) l1[j] = s;
  }
  __syncthreads();

  // softmax over 4000 (fixed-order tree)
  float lm = -3.4e38f;
  for (int j = tid; j < 4000; j += 256) lm = fmaxf(lm, l1[j]);
  red[tid] = lm; __syncthreads();
  for (int sft = 128; sft > 0; sft >>= 1) {
    if (tid < sft) red[tid] = fmaxf(red[tid], red[tid + sft]);
    __syncthreads();
  }
  const float mx1 = red[0];
  __syncthreads();
  float ls = 0.f;
  for (int j = tid; j < 4000; j += 256) {
    const float e = __expf(l1[j] - mx1);
    l1[j] = e;
    ls += e;
  }
  red[tid] = ls; __syncthreads();
  for (int sft = 128; sft > 0; sft >>= 1) {
    if (tid < sft) red[tid] += red[tid + sft];
    __syncthreads();
  }
  const float sum1 = red[0];
  __syncthreads();
  for (int j = tid; j < 4000; j += 256) l1[j] /= sum1;
  __syncthreads();

  // fact1 (coalesced over c)
  {
    float s = 0.f;
    for (int j = 0; j < 4000; ++j) s += l1[j] * dm[(size_t)j * 256 + tid];
    f1[tid] = s;
  }

  // visit weights over 63 history keys
  if (tid < 63) {
    float s = 0.f;
    for (int d = 0; d < 256; ++d) s += qs[d] * feat[tid * 256 + d];
    l2[tid] = s;
  }
  __syncthreads();
  if (tid == 0) {
    float mx = -3.4e38f;
    for (int t = 0; t < 63; ++t) mx = fmaxf(mx, l2[t]);
    float se = 0.f;
    for (int t = 0; t < 63; ++t) se += __expf(l2[t] - mx);
    for (int t = 0; t < 63; ++t) l2[t] = __expf(l2[t] - mx) / se;
  }
  __syncthreads();

  // agg[m] = sum_t visit_w[t] * [m in codes of visit t]  (set semantics)
  for (int m = tid; m < 4000; m += 256) {
    float a = 0.f;
    for (int t = 0; t < 63; ++t) {
      bool found = false;
      for (int p = 0; p < 16; ++p)
        if (cds[t * 16 + p] == m) { found = true; break; }
      if (found) a += l2[t];
    }
    agg[m] = a;
  }
  __syncthreads();

  // fact2 (coalesced over c)
  {
    float s = 0.f;
    for (int m = 0; m < 4000; ++m) s += agg[m] * dm[(size_t)m * 256 + tid];
    f2[tid] = s;
  }
  __syncthreads();

  // hidden = relu([q,f1,f2] @ w1 + b1), two outputs per thread
  const int o0 = tid, o1 = tid + 256;
  float a0 = b1[o0], a1 = b1[o1];
  for (int d = 0; d < 256; ++d) {
    const float h = qs[d];
    a0 += h * w1[d * 512 + o0];
    a1 += h * w1[d * 512 + o1];
  }
  for (int d = 0; d < 256; ++d) {
    const float h = f1[d];
    a0 += h * w1[(256 + d) * 512 + o0];
    a1 += h * w1[(256 + d) * 512 + o1];
  }
  for (int d = 0; d < 256; ++d) {
    const float h = f2[d];
    a0 += h * w1[(512 + d) * 512 + o0];
    a1 += h * w1[(512 + d) * 512 + o1];
  }
  hidden[o0] = fmaxf(a0, 0.f);
  hidden[o1] = fmaxf(a1, 0.f);
}

// ---------------------------------------------------------------------------
// K5b: result = hidden @ out_w2 + out_b2 ; sp = sigmoid(result)
// grid(16), block(256)
// ---------------------------------------------------------------------------
__global__ __launch_bounds__(256)
void out_proj(const float* __restrict__ hidden,
              const float* __restrict__ w2,
              const float* __restrict__ b2,
              float* __restrict__ outv,
              float* __restrict__ sp) {
  __shared__ float hs[512];
  hs[threadIdx.x]       = hidden[threadIdx.x];
  hs[threadIdx.x + 256] = hidden[threadIdx.x + 256];
  __syncthreads();
  const int j = blockIdx.x * 256 + threadIdx.x;
  if (j < 4000) {
    float s = b2[j];
#pragma unroll 4
    for (int d = 0; d < 512; ++d) s += hs[d] * w2[(size_t)d * 4000 + j];
    outv[j] = s;
    sp[j] = 1.0f / (1.0f + __expf(-s));
  }
}

// ---------------------------------------------------------------------------
// K6: per-row partials of sp^T * DDI * sp   (one wave -> 4 rows)
// ---------------------------------------------------------------------------
__global__ __launch_bounds__(256)
void ddi_rows(const float* __restrict__ ddi,
              const float* __restrict__ sp,
              float* __restrict__ part) {
  const int gw   = (blockIdx.x * blockDim.x + threadIdx.x) >> 5;
  const int lane = threadIdx.x & 31;
#pragma unroll
  for (int r = 0; r < 4; ++r) {
    const int i = gw * 4 + r;
    float s = 0.f;
    for (int j = lane; j < 4000; j += 32) s += ddi[(size_t)i * 4000 + j] * sp[j];
    s = wred_sum(s);
    if (lane == 0) part[i] = sp[i] * s;
  }
}

// ---------------------------------------------------------------------------
// K7: fixed-order final reduction -> d_out[4000]
// ---------------------------------------------------------------------------
__global__ __launch_bounds__(256)
void final_reduce(const float* __restrict__ part, float* __restrict__ outv) {
  __shared__ float red[256];
  const int tid = threadIdx.x;
  float s = 0.f;
  for (int t = tid; t < 4000; t += 256) s += part[t];
  red[tid] = s;
  __syncthreads();
  for (int sft = 128; sft > 0; sft >>= 1) {
    if (tid < sft) red[tid] += red[tid + sft];
    __syncthreads();
  }
  if (tid == 0) outv[4000] = 0.0005f * red[0];
}

// ---------------------------------------------------------------------------
extern "C" void kernel_launch(void* const* d_in, const int* in_sizes, int n_in,
                              void* d_out, int out_size, void* d_ws, size_t ws_size,
                              hipStream_t stream) {
  const int*   proc_codes = (const int*)d_in[0];
  const int*   med_codes  = (const int*)d_in[1];
  const float* emb_table  = (const float*)d_in[2];
  const float* conv_w     = (const float*)d_in[3];
  const float* conv_b     = (const float*)d_in[4];
  const float* wq         = (const float*)d_in[5];
  const float* wk         = (const float*)d_in[6];
  const float* wv         = (const float*)d_in[7];
  const float* wr         = (const float*)d_in[8];
  const float* alpha      = (const float*)d_in[9];
  const float* beta       = (const float*)d_in[10];
  const float* ehr_adj    = (const float*)d_in[11];
  const float* ddi_adj    = (const float*)d_in[12];
  const float* ddi_raw    = (const float*)d_in[13];
  const float* ehr_w1     = (const float*)d_in[14];
  const float* ehr_b1     = (const float*)d_in[15];
  const float* ehr_w2     = (const float*)d_in[16];
  const float* ehr_b2     = (const float*)d_in[17];
  const float* ddi_w1     = (const float*)d_in[18];
  const float* ddi_b1     = (const float*)d_in[19];
  const float* ddi_w2     = (const float*)d_in[20];
  const float* ddi_b2     = (const float*)d_in[21];
  const float* inter      = (const float*)d_in[22];
  const float* out_w1     = (const float*)d_in[23];
  const float* out_b1     = (const float*)d_in[24];
  const float* out_w2     = (const float*)d_in[25];
  const float* out_b2     = (const float*)d_in[26];

  float* W      = (float*)d_ws;
  float* i2     = W;                     // 64*128
  float* q      = i2 + 8192;             // 64*256
  float* k      = q + 16384;
  float* v      = k + 16384;
  float* gk     = v + 16384;             // 256
  float* feat   = gk + 256;              // 64*256
  float* H      = feat + 16384;          // 4000*256
  float* HW2    = H + 1024000;           // 4000*256
  float* DM     = HW2 + 1024000;         // 4000*256
  float* hidden = DM + 1024000;          // 512
  float* sp     = hidden + 512;          // 4000
  float* part   = sp + 4000;             // 4000
  __bf16* Bpack = (__bf16*)(part + 4000);            // 4000*256 bf16 (2 MB)
  __bf16* Abf   = (__bf16*)(part + 4000 + 512000);   // 4000*4000 bf16 (32 MB)
  __bf16* Hbf   = (__bf16*)(part + 4000 + 512000 + 8000000); // 4000*256 bf16 (2 MB)
  float* outv   = (float*)d_out;         // 4000 result + 1 batch_neg

  // --- front end ---
  embed_conv_kernel<<<64, 128, 0, stream>>>(proc_codes, emb_table, conv_w, conv_b, i2);
  qkv_kernel<<<192, 256, 0, stream>>>(i2, wq, wk, wv, q, k, v);
  fastformer_gates<<<1, 256, 0, stream>>>(q, k, alpha, beta, gk);
  feat_combine<<<64, 256, 0, stream>>>(q, v, gk, wr, feat);

  // --- GCNs via bf16 WMMA (1000 wave-tiles -> 125 blocks of 8 waves) ---
  // EHR branch
  convert_bf16<<<15625, 256, 0, stream>>>(ehr_adj, 16000000, Abf);
  pack_b_bf16<<<4000, 256, 0, stream>>>(ehr_w1, 4000, Bpack);
  gemm_wmma_n256<<<125, 256, 0, stream>>>(Abf, 4000, Bpack, H,   4000, 4000, ehr_b1, 1, nullptr, 1.0f, 0);
  convert_bf16<<<1000, 256, 0, stream>>>(H, 1024000, Hbf);
  pack_b_bf16<<<256, 256, 0, stream>>>(ehr_w2, 256, Bpack);
  gemm_wmma_n256<<<125, 256, 0, stream>>>(Hbf, 256, Bpack, HW2, 4000,  256, nullptr, 0, nullptr, 1.0f, 0);
  pack_b_bf16<<<4000, 256, 0, stream>>>(HW2, 4000, Bpack);
  gemm_wmma_n256<<<125, 256, 0, stream>>>(Abf, 4000, Bpack, DM,  4000, 4000, ehr_b2, 0, nullptr, 1.0f, 0);
  // DDI branch
  convert_bf16<<<15625, 256, 0, stream>>>(ddi_adj, 16000000, Abf);
  pack_b_bf16<<<4000, 256, 0, stream>>>(ddi_w1, 4000, Bpack);
  gemm_wmma_n256<<<125, 256, 0, stream>>>(Abf, 4000, Bpack, H,   4000, 4000, ddi_b1, 1, nullptr, 1.0f, 0);
  convert_bf16<<<1000, 256, 0, stream>>>(H, 1024000, Hbf);
  pack_b_bf16<<<256, 256, 0, stream>>>(ddi_w2, 256, Bpack);
  gemm_wmma_n256<<<125, 256, 0, stream>>>(Hbf, 256, Bpack, HW2, 4000,  256, nullptr, 0, nullptr, 1.0f, 0);
  pack_b_bf16<<<4000, 256, 0, stream>>>(HW2, 4000, Bpack);
  // DM -= inter * (ddi_adj @ HW2 + b2)
  gemm_wmma_n256<<<125, 256, 0, stream>>>(Abf, 4000, Bpack, DM,  4000, 4000, ddi_b2, 0, inter, -1.0f, 1);

  // --- decoder + DDI penalty ---
  decode_head<<<1, 256, 0, stream>>>(feat, DM, med_codes, out_w1, out_b1, hidden);
  out_proj<<<16, 256, 0, stream>>>(hidden, out_w2, out_b2, outv, sp);
  ddi_rows<<<125, 256, 0, stream>>>(ddi_raw, sp, part);
  final_reduce<<<1, 256, 0, stream>>>(part, outv);
}
